// WindowAttention3D_58617713655995
// MI455X (gfx1250) — matrix operand order repaired and checked by reading, not compiled, not run on hardware
//
#include <hip/hip_runtime.h>
#include <hip/hip_bf16.h>

typedef _Float16 v16h __attribute__((ext_vector_type(16)));
typedef _Float16 v8h  __attribute__((ext_vector_type(8)));
typedef _Float16 v4h  __attribute__((ext_vector_type(4)));
typedef float    v8f  __attribute__((ext_vector_type(8)));

#define WMMA_F32_F16(a, b, c) \
  __builtin_amdgcn_wmma_f32_16x16x32_f16(false, (a), false, (b), (short)0, (c), false, false)

static constexpr int Nseq = 343;   // tokens per window (7*7*7)
static constexpr int NP   = 352;   // padded to 22 tiles of 16
static constexpr int C    = 96;
static constexpr int HD   = 32;
static constexpr int C3   = 288;
static constexpr int QSTRIDE = NP * HD;      // 11264 halves per head plane
static constexpr int BIASN   = Nseq * Nseq;  // 117649

// LDS layout (in halves)
static constexpr int OFF_X = 0;          // sX / sOut : 352*96 = 33792
static constexpr int OFF_Q = 33792;      // sQ  [3][352][32]
static constexpr int OFF_K = 67584;      // sKt [3][32][352]
static constexpr int OFF_V = 101376;     // sV  [3][352][32]
static constexpr int OFF_P = 135168;     // sP  [8 waves][16*32]
static constexpr int LDS_HALVES = 139264; // 278528 bytes

static __device__ __forceinline__ v16h ld16(const _Float16* p0, const _Float16* p1) {
  v8h a = *(const v8h*)p0;
  v8h b = *(const v8h*)p1;
  return __builtin_shufflevector(a, b, 0, 1, 2, 3, 4, 5, 6, 7, 8, 9, 10, 11, 12, 13, 14, 15);
}

// ---- prep kernel 1: convert w_qkv (96x288) and w_proj (96x96) to f16 in ws
__global__ __launch_bounds__(256) void prep_weights_kernel(
    const float* __restrict__ wqkv, const float* __restrict__ wproj,
    _Float16* __restrict__ wq16, _Float16* __restrict__ wp16) {
  const int i = blockIdx.x * 256 + threadIdx.x;
  if (i < C * C3) wq16[i] = (_Float16)wqkv[i];
  if (i < C * C)  wp16[i] = (_Float16)wproj[i];
}

// ---- prep kernel 2: gather relative-position bias -> biasw[h][i][j]
__global__ __launch_bounds__(256) void prep_bias_kernel(
    const float* __restrict__ table, float* __restrict__ biasw) {
  const int ij = blockIdx.x * 256 + threadIdx.x;
  if (ij >= BIASN) return;
  const int i = ij / Nseq, j = ij % Nseq;
  const int iw = i / 49, ih = (i / 7) % 7, iz = i % 7;
  const int jw = j / 49, jh = (j / 7) % 7, jz = j % 7;
  const int idx = (iw - jw + 6) * 169 + (ih - jh + 6) * 13 + (iz - jz + 6);
#pragma unroll
  for (int h = 0; h < 3; ++h) biasw[h * BIASN + ij] = table[idx * 3 + h];
}

// ---- main fused window-attention kernel: one block per (batch,window)
__global__ __launch_bounds__(256) void swin_attn_kernel(
    const float* __restrict__ x, const float* __restrict__ mask,
    const float* __restrict__ b_qkv, const float* __restrict__ b_proj,
    const _Float16* __restrict__ wqkv16, const _Float16* __restrict__ wproj16,
    const float* __restrict__ biasw, float* __restrict__ out) {
  extern __shared__ __align__(16) _Float16 smem[];
  _Float16* sX  = smem + OFF_X;   // staged x (f16), later reused as attention output
  _Float16* sQ  = smem + OFF_Q;
  _Float16* sKt = smem + OFF_K;
  _Float16* sV  = smem + OFF_V;
  _Float16* sP  = smem + OFF_P;

  const int tid  = threadIdx.x;
  const int lane = tid & 31;
  const int wave = tid >> 5;
  const int hi   = (lane >> 4) & 1;   // which half of the wave
  const int ln   = lane & 15;
  const int b    = blockIdx.x;

  // ---------- Phase 0: stage x (fp32 -> f16), zero pad rows ----------
  const float* xb = x + (size_t)b * (Nseq * C);
  for (int i = tid; i < (Nseq * C) / 4; i += 256) {
    const float4 f = ((const float4*)xb)[i];
    v4h h;
    h[0] = (_Float16)f.x; h[1] = (_Float16)f.y;
    h[2] = (_Float16)f.z; h[3] = (_Float16)f.w;
    *(v4h*)(sX + i * 4) = h;
  }
  for (int i = tid; i < (NP - Nseq) * C; i += 256) sX[Nseq * C + i] = (_Float16)0.f;
  __syncthreads();

  // ---------- Phase 1: QKV = x @ Wqkv + b (K stored transposed) ----------
  const float scaleq = 0.17677669529663687f;  // 32^-0.5
  for (int t = wave; t < 22 * 18; t += 8) {
    const int m = t / 18, n = t % 18;
    const int row = m * 16 + ln;
    v8f acc = {};
#pragma unroll
    for (int k = 0; k < 3; ++k) {
      const _Float16* pa = sX + row * C + k * 32 + (hi ? 8 : 0);
      const v16h A = ld16(pa, pa + 16);
      const _Float16* pb = wqkv16 + (k * 32) * C3 + n * 16 + ln;
      v16h B;
#pragma unroll
      for (int s = 0; s < 16; ++s) B[s] = pb[((s & 7) + (hi ? 8 : 0) + ((s >> 3) << 4)) * C3];
      acc = WMMA_F32_F16(A, B, acc);
    }
    const float bq = b_qkv[n * 16 + ln];
    if (n < 6) {                       // Q (scaled)
      const int h = n >> 1, doff = (n & 1) * 16;
      _Float16* dst = sQ + h * QSTRIDE;
#pragma unroll
      for (int r = 0; r < 8; ++r) {
        const int ri = m * 16 + r + (hi ? 8 : 0);
        dst[ri * HD + doff + ln] = (_Float16)((acc[r] + bq) * scaleq);
      }
    } else if (n < 12) {               // K, transposed -> [hd][N]
      const int nn = n - 6, h = nn >> 1, doff = (nn & 1) * 16;
      _Float16* dst = sKt + h * QSTRIDE;
#pragma unroll
      for (int r = 0; r < 8; ++r) {
        const int ri = m * 16 + r + (hi ? 8 : 0);
        dst[(doff + ln) * NP + ri] = (_Float16)(acc[r] + bq);
      }
    } else {                           // V
      const int nn = n - 12, h = nn >> 1, doff = (nn & 1) * 16;
      _Float16* dst = sV + h * QSTRIDE;
#pragma unroll
      for (int r = 0; r < 8; ++r) {
        const int ri = m * 16 + r + (hi ? 8 : 0);
        dst[ri * HD + doff + ln] = (_Float16)(acc[r] + bq);
      }
    }
  }
  __syncthreads();

  // ---------- Phase 2: flash attention per (row-tile, head) ----------
  const float* maskw = mask + (size_t)(b & 31) * BIASN;
  _Float16* sPw = sP + wave * 512;
  for (int p = wave; p < 22 * 3; p += 8) {
    const int m = p / 3, h = p % 3;
    const int mb = m * 16;
    const _Float16* qrow = sQ + h * QSTRIDE + (mb + ln) * HD + (hi ? 8 : 0);
    const v16h AQ = ld16(qrow, qrow + 16);
    float mrun[8], lrun[8];
    v8f o0 = {}, o1 = {};
#pragma unroll
    for (int r = 0; r < 8; ++r) { mrun[r] = -3.0e38f; lrun[r] = 0.f; }

    for (int jj = 0; jj < 11; ++jj) {
#pragma unroll
      for (int t2 = 0; t2 < 2; ++t2) {
        const int jb = (jj * 2 + t2) * 16;
        const _Float16* kp = sKt + h * QSTRIDE + jb + ln;
        v16h BK;
#pragma unroll
        for (int s = 0; s < 16; ++s) BK[s] = kp[((s & 7) + (hi ? 8 : 0) + ((s >> 3) << 4)) * NP];
        v8f z = {};
        v8f s = WMMA_F32_F16(AQ, BK, z);
        const int col = jb + ln;
        const bool cv = (col < Nseq);
#pragma unroll
        for (int r = 0; r < 8; ++r) {
          const int ri = mb + r + (hi ? 8 : 0);
          float sv = s[r];
          if (cv && ri < Nseq)
            sv += biasw[h * BIASN + ri * Nseq + col] + maskw[ri * Nseq + col];
          if (!cv) sv = -1.0e30f;
          // half-wave row max (row lives on 16 lanes of one half)
          float tm = sv;
          tm = fmaxf(tm, __shfl_xor(tm, 8, 16));
          tm = fmaxf(tm, __shfl_xor(tm, 4, 16));
          tm = fmaxf(tm, __shfl_xor(tm, 2, 16));
          tm = fmaxf(tm, __shfl_xor(tm, 1, 16));
          const float nm = fmaxf(mrun[r], tm);
          const float sc = __expf(mrun[r] - nm);
          const float pv = __expf(sv - nm);
          float rs = pv;
          rs += __shfl_xor(rs, 8, 16);
          rs += __shfl_xor(rs, 4, 16);
          rs += __shfl_xor(rs, 2, 16);
          rs += __shfl_xor(rs, 1, 16);
          lrun[r] = lrun[r] * sc + rs;
          mrun[r] = nm;
          o0[r] *= sc;
          o1[r] *= sc;
          // stage P (C-layout -> row-major 16x32) for the PV matmul
          sPw[(r + (hi ? 8 : 0)) * 32 + t2 * 16 + ln] = (_Float16)pv;
        }
      }
      // O += P(16x32) @ V(32x32)
      const _Float16* pp = sPw + ln * 32 + (hi ? 8 : 0);
      const v16h AP = ld16(pp, pp + 16);
      const _Float16* vp = sV + h * QSTRIDE + (jj * 32) * HD + ln;
      v16h BV0, BV1;
#pragma unroll
      for (int s = 0; s < 16; ++s) {
        const int kk = (s & 7) + (hi ? 8 : 0) + ((s >> 3) << 4);
        BV0[s] = vp[kk * HD];
        BV1[s] = vp[kk * HD + 16];
      }
      o0 = WMMA_F32_F16(AP, BV0, o0);
      o1 = WMMA_F32_F16(AP, BV1, o1);
    }
    // normalize and write attention output (f16) into sX (reused as sOut)
#pragma unroll
    for (int r = 0; r < 8; ++r) {
      const float inv = 1.0f / lrun[r];
      const int ri = mb + r + (hi ? 8 : 0);
      sX[ri * C + h * HD + ln]      = (_Float16)(o0[r] * inv);
      sX[ri * C + h * HD + 16 + ln] = (_Float16)(o1[r] * inv);
    }
  }
  __syncthreads();

  // ---------- Phase 3: out = attn_out @ Wproj + b_proj ----------
  float* ob = out + (size_t)b * (Nseq * C);
  for (int t = wave; t < 22 * 6; t += 8) {
    const int m = t / 6, n = t % 6;
    const int row = m * 16 + ln;
    v8f acc = {};
#pragma unroll
    for (int k = 0; k < 3; ++k) {
      const _Float16* pa = sX + row * C + k * 32 + (hi ? 8 : 0);
      const v16h A = ld16(pa, pa + 16);
      const _Float16* pb = wproj16 + (k * 32) * C + n * 16 + ln;
      v16h B;
#pragma unroll
      for (int s = 0; s < 16; ++s) B[s] = pb[((s & 7) + (hi ? 8 : 0) + ((s >> 3) << 4)) * C];
      acc = WMMA_F32_F16(A, B, acc);
    }
    const float bp = b_proj[n * 16 + ln];
#pragma unroll
    for (int r = 0; r < 8; ++r) {
      const int ri = m * 16 + r + (hi ? 8 : 0);
      if (ri < Nseq) ob[ri * C + n * 16 + ln] = acc[r] + bp;
    }
  }
}

extern "C" void kernel_launch(void* const* d_in, const int* in_sizes, int n_in,
                              void* d_out, int out_size, void* d_ws, size_t ws_size,
                              hipStream_t stream) {
  const float* x     = (const float*)d_in[0];
  const float* mask  = (const float*)d_in[1];
  const float* wqkv  = (const float*)d_in[2];
  const float* bqkv  = (const float*)d_in[3];
  const float* btab  = (const float*)d_in[4];
  const float* wproj = (const float*)d_in[5];
  const float* bproj = (const float*)d_in[6];
  float* out = (float*)d_out;

  char* ws = (char*)d_ws;
  _Float16* wq16  = (_Float16*)ws;                 // 55296 halves
  _Float16* wp16  = (_Float16*)(ws + 110592);      // 9216 halves
  float*    biasw = (float*)(ws + 129024);         // 3*117649 floats

  prep_weights_kernel<<<(C * C3 + 255) / 256, 256, 0, stream>>>(wqkv, wproj, wq16, wp16);
  prep_bias_kernel<<<(BIASN + 255) / 256, 256, 0, stream>>>(btab, biasw);

  const size_t lds_bytes = (size_t)LDS_HALVES * sizeof(_Float16);  // 278528 B (<320KB WGP LDS)
  swin_attn_kernel<<<256, 256, lds_bytes, stream>>>(x, mask, bqkv, bproj, wq16, wp16, biasw, out);

  (void)in_sizes; (void)n_in; (void)out_size; (void)ws_size;
}
